// TagNCModel_BGBBase_17437567221947
// MI455X (gfx1250) — matrix operand order repaired and checked by reading, not compiled, not run on hardware
//
#include <hip/hip_runtime.h>
#include <hip/hip_bf16.h>

typedef __attribute__((ext_vector_type(16))) _Float16 v16h;
typedef __attribute__((ext_vector_type(8)))  _Float16 v8h;
typedef __attribute__((ext_vector_type(8)))  float    v8f;

#define F_IN 128
#define HID  128
#define LDH  128   // row stride (floats/halfs) of all activation buffers

// ---------------------------------------------------------------------------
// Degree kernels: deg starts at 1.0 (self loop), +1 per incoming edge, rsqrt.
// ---------------------------------------------------------------------------
__global__ void k_deg_init(float* deg, int N) {
    int i = blockIdx.x * blockDim.x + threadIdx.x;
    if (i < N) deg[i] = 1.0f;
}

__global__ void k_deg_count(const int* __restrict__ col, float* deg, int E) {
    int e = blockIdx.x * blockDim.x + threadIdx.x;
    if (e < E) atomicAdd(&deg[col[e]], 1.0f);
}

__global__ void k_deg_rsqrt(float* deg, int N) {
    int i = blockIdx.x * blockDim.x + threadIdx.x;
    if (i < N) deg[i] = rsqrtf(fmaxf(deg[i], 1e-12f));
}

// ---------------------------------------------------------------------------
// fp32 -> f16 convert (optionally fused ReLU). Feeds the WMMA GEMM.
// ---------------------------------------------------------------------------
__global__ void k_convert_f16(const float* __restrict__ in, _Float16* __restrict__ out,
                              int n, int relu) {
    int i = blockIdx.x * blockDim.x + threadIdx.x;
    if (i < n) {
        float v = in[i];
        if (relu) v = fmaxf(v, 0.0f);
        out[i] = (_Float16)v;
    }
}

// ---------------------------------------------------------------------------
// WMMA GEMM: Y[N x LDH] = Xh[N x 128](f16) @ W[128 x Wo](f32, converted).
// Block = 128 threads = 4 waves; each wave owns a 16-row stripe, all column
// tiles. W is pre-swizzled into LDS in B-fragment order (ISA 7.12.2) so each
// lane reads its 16 halfs with contiguous ds_load_b128.
//   B-frag: lane holds col = nt*16 + (lane&15); K = kt*32 + (lane>=16?8:0)+i+(i>=8?8:0)
//   A-frag: lane holds row = (lane&15); K runs [base,base+8) and
//           [base+16,base+24), base = kt*32 + (lane>=16?8:0)
//   C/D   : vgpr r -> row = r + (lane>=16?8:0); col = nt*16 + (lane&15)
// ---------------------------------------------------------------------------
template <int HO_TILES>
__global__ void k_gemm_wmma(const _Float16* __restrict__ Xh,
                            const float* __restrict__ W,
                            float* __restrict__ Y,
                            int N, int Wo) {
    __shared__ _Float16 wlds[4 * HO_TILES * 32 * 16];

    const int tid  = threadIdx.x;
    const int lane = tid & 31;
    const int wave = tid >> 5;

    // ---- pre-swizzle W (f32 -> f16) into B-fragment order -----------------
    const int total = 4 * HO_TILES * 32 * 16;
    for (int t = tid; t < total; t += blockDim.x) {
        int i    = t & 15;
        int ln   = (t >> 4) & 31;
        int ntk  = t >> 9;             // kt*HO_TILES + nt
        int nt   = ntk % HO_TILES;
        int kt   = ntk / HO_TILES;
        int K    = kt * 32 + ((ln >> 4) << 3) + i + ((i >> 3) << 3);   // 0..127
        int colg = nt * 16 + (ln & 15);
        wlds[t] = (colg < Wo) ? (_Float16)W[K * Wo + colg] : (_Float16)0.0f;
    }
    __syncthreads();

    const int rowTile = blockIdx.x * 64 + wave * 16;
    const int myRow   = rowTile + (lane & 15);
    const bool rowOk  = (myRow < N);

    v8f acc[HO_TILES];
#pragma unroll
    for (int nt = 0; nt < HO_TILES; ++nt) acc[nt] = {};

#pragma unroll
    for (int kt = 0; kt < 4; ++kt) {
        // ---- A fragment: two contiguous runs of 8 f16 ---------------------
        const int kbase = kt * 32 + ((lane >> 4) << 3);   // +8 for upper lanes
        v8h r0 = {}, r1 = {};
        if (rowOk) {
            const _Float16* ap = Xh + (size_t)myRow * LDH + kbase;
            r0 = *reinterpret_cast<const v8h*>(ap);        // K = base..base+7
            r1 = *reinterpret_cast<const v8h*>(ap + 16);   // K = base+16..+23
        }
        v16h a;
#pragma unroll
        for (int j = 0; j < 8; ++j) { a[j] = r0[j]; a[8 + j] = r1[j]; }

#pragma unroll
        for (int nt = 0; nt < HO_TILES; ++nt) {
            const _Float16* bp = &wlds[((kt * HO_TILES + nt) * 32 + lane) * 16];
            v8h b0 = *reinterpret_cast<const v8h*>(bp);
            v8h b1 = *reinterpret_cast<const v8h*>(bp + 8);
            v16h b;
#pragma unroll
            for (int j = 0; j < 8; ++j) { b[j] = b0[j]; b[8 + j] = b1[j]; }

            acc[nt] = __builtin_amdgcn_wmma_f32_16x16x32_f16(
                false, a, false, b, (short)0, acc[nt], false, false);
        }
    }

    // ---- store D ----------------------------------------------------------
    const int rOff = (lane >> 4) << 3;   // lanes 16..31 -> rows +8
    const int cOff = lane & 15;
#pragma unroll
    for (int nt = 0; nt < HO_TILES; ++nt) {
#pragma unroll
        for (int r = 0; r < 8; ++r) {
            int row = rowTile + r + rOff;
            if (row < N)
                Y[(size_t)row * LDH + nt * 16 + cOff] = acc[nt][r];
        }
    }
}

// ---------------------------------------------------------------------------
// Aggregation init: out[i,c] = b[c] + hw[i,c] * dis[i]^2   (self-loop + bias)
// Processes HoPad columns (stride LDH).
// ---------------------------------------------------------------------------
__global__ void k_agg_init(const float* __restrict__ hw, const float* __restrict__ dis,
                           const float* __restrict__ bias, float* __restrict__ out,
                           int N, int HoPad, int Ho) {
    int idx = blockIdx.x * blockDim.x + threadIdx.x;
    if (idx >= N * HoPad) return;
    int i = idx / HoPad;
    int c = idx - i * HoPad;
    float d  = dis[i];
    float bc = (c < Ho) ? bias[c] : 0.0f;
    out[(size_t)i * LDH + c] = bc + hw[(size_t)i * LDH + c] * d * d;
}

// ---------------------------------------------------------------------------
// Edge scatter: one wave32 per edge. Lanes < nchunk each own a float4 chunk
// (4 channels). out[col] += h[row] * dis[row]*dis[col] via global f32 atomics.
// ---------------------------------------------------------------------------
__global__ void k_agg_edges(const int* __restrict__ row, const int* __restrict__ col,
                            const float4* __restrict__ h4, const float* __restrict__ dis,
                            float* __restrict__ out, int E, int nchunk) {
    const int lane  = threadIdx.x & 31;
    const int waveG = (blockIdx.x * blockDim.x + threadIdx.x) >> 5;
    const int nWave = (gridDim.x * blockDim.x) >> 5;

    for (int e = waveG; e < E; e += nWave) {
        if (e + 4 * nWave < E) {
            __builtin_prefetch(&row[e + 4 * nWave], 0, 0);
            __builtin_prefetch(&col[e + 4 * nWave], 0, 0);
        }
        int r = row[e];
        int c = col[e];
        float nrm = dis[r] * dis[c];
        if (lane < nchunk) {
            float4 v = h4[(size_t)r * (LDH / 4) + lane];
            float* op = out + (size_t)c * LDH + lane * 4;
            atomicAdd(op + 0, v.x * nrm);
            atomicAdd(op + 1, v.y * nrm);
            atomicAdd(op + 2, v.z * nrm);
            atomicAdd(op + 3, v.w * nrm);
        }
    }
}

// ---------------------------------------------------------------------------
// Log-softmax over C classes, one wave per node. Input stride LDH, out stride C.
// ---------------------------------------------------------------------------
__global__ void k_log_softmax(const float* __restrict__ in, float* __restrict__ out,
                              int N, int C) {
    const int lane = threadIdx.x & 31;
    const int node = (blockIdx.x * blockDim.x + threadIdx.x) >> 5;
    if (node >= N) return;

    const float* rp = in + (size_t)node * LDH;
    float v0 = (lane      < C) ? rp[lane]      : -INFINITY;
    float v1 = (lane + 32 < C) ? rp[lane + 32] : -INFINITY;

    float m = fmaxf(v0, v1);
#pragma unroll
    for (int off = 16; off > 0; off >>= 1) m = fmaxf(m, __shfl_xor(m, off, 32));

    float s = ((lane < C) ? expf(v0 - m) : 0.0f) +
              ((lane + 32 < C) ? expf(v1 - m) : 0.0f);
#pragma unroll
    for (int off = 16; off > 0; off >>= 1) s += __shfl_xor(s, off, 32);

    float ls = logf(s);
    float* op = out + (size_t)node * C;
    if (lane < C)      op[lane]      = v0 - m - ls;
    if (lane + 32 < C) op[lane + 32] = v1 - m - ls;
}

// ---------------------------------------------------------------------------
// Host launcher
// ---------------------------------------------------------------------------
static inline int cdiv(long a, long b) { return (int)((a + b - 1) / b); }

extern "C" void kernel_launch(void* const* d_in, const int* in_sizes, int n_in,
                              void* d_out, int out_size, void* d_ws, size_t ws_size,
                              hipStream_t stream) {
    const float* x  = (const float*)d_in[0];
    const int*   ei = (const int*)  d_in[1];
    const float* W1 = (const float*)d_in[2];
    const float* b1 = (const float*)d_in[3];
    const float* W2 = (const float*)d_in[4];
    const float* b2 = (const float*)d_in[5];
    const float* W3 = (const float*)d_in[6];
    const float* b3 = (const float*)d_in[7];
    float* outp = (float*)d_out;

    const int N = in_sizes[0] / F_IN;
    const int E = in_sizes[1] / 2;
    const int C = in_sizes[7];            // 40
    const int CPAD = ((C + 15) / 16) * 16;  // 48

    const int* rowI = ei;        // edge_index[0]
    const int* colI = ei + E;    // edge_index[1]

    // ---- workspace carve-up (256B aligned) -------------------------------
    char* ws = (char*)d_ws;
    size_t off = 0;
    auto alloc = [&](size_t bytes) { void* p = ws + off; off = (off + bytes + 255) & ~(size_t)255; return p; };
    float*    dis  = (float*)   alloc((size_t)N * 4);
    _Float16* hf16 = (_Float16*)alloc((size_t)N * LDH * 2);
    float*    bufA = (float*)   alloc((size_t)N * LDH * 4);
    float*    bufB = (float*)   alloc((size_t)N * LDH * 4);

    const int T = 256;
    const int edgeBlocks = 8192;

    // ---- degrees ---------------------------------------------------------
    k_deg_init <<<cdiv(N, T), T, 0, stream>>>(dis, N);
    k_deg_count<<<cdiv(E, T), T, 0, stream>>>(colI, dis, E);
    k_deg_rsqrt<<<cdiv(N, T), T, 0, stream>>>(dis, N);

    const int nElem = N * LDH;
    const int gemmBlocks = cdiv(N, 64);

    // ---- layer 1 ---------------------------------------------------------
    k_convert_f16<<<cdiv(nElem, T), T, 0, stream>>>(x, hf16, nElem, 0);
    k_gemm_wmma<8><<<gemmBlocks, 128, 0, stream>>>(hf16, W1, bufA, N, HID);
    k_agg_init <<<cdiv((long)N * HID, T), T, 0, stream>>>(bufA, dis, b1, bufB, N, HID, HID);
    k_agg_edges<<<edgeBlocks, T, 0, stream>>>(rowI, colI, (const float4*)bufA, dis, bufB, E, HID / 4);

    // ---- layer 2 (ReLU fused into convert) -------------------------------
    k_convert_f16<<<cdiv(nElem, T), T, 0, stream>>>(bufB, hf16, nElem, 1);
    k_gemm_wmma<8><<<gemmBlocks, 128, 0, stream>>>(hf16, W2, bufA, N, HID);
    k_agg_init <<<cdiv((long)N * HID, T), T, 0, stream>>>(bufA, dis, b2, bufB, N, HID, HID);
    k_agg_edges<<<edgeBlocks, T, 0, stream>>>(rowI, colI, (const float4*)bufA, dis, bufB, E, HID / 4);

    // ---- layer 3 (C=40 padded to 48 cols, zero weights beyond 40) --------
    k_convert_f16<<<cdiv(nElem, T), T, 0, stream>>>(bufB, hf16, nElem, 1);
    k_gemm_wmma<3><<<gemmBlocks, 128, 0, stream>>>(hf16, W3, bufA, N, C);
    k_agg_init <<<cdiv((long)N * CPAD, T), T, 0, stream>>>(bufA, dis, b3, bufB, N, CPAD, C);
    k_agg_edges<<<edgeBlocks, T, 0, stream>>>(rowI, colI, (const float4*)bufA, dis, bufB, E, CPAD / 4);

    // ---- log_softmax -> d_out -------------------------------------------
    k_log_softmax<<<cdiv((long)N * 32, T), T, 0, stream>>>(bufB, outp, N, C);
}
